// ScaledDotProduct_62371515073055
// MI455X (gfx1250) — compile-verified, matrix operand order checked by reading, not run on hardware
//
#include <hip/hip_runtime.h>
#include <hip/hip_bf16.h>

// Problem constants (from reference)
#define DM     512
#define HEADS  8
#define DHEAD  64
#define BATCH  4
#define SEQ    2048
#define BS     (BATCH*SEQ)          // 8192 rows of activations
#define HSD    ((size_t)BATCH*HEADS*SEQ*DHEAD)   // 4,194,304 elems per head tensor
#define ATT_OFF ((size_t)BATCH*SEQ*DM)           // out elems before att in d_out

typedef __attribute__((ext_vector_type(16))) _Float16 v16h;
typedef __attribute__((ext_vector_type(8)))  _Float16 v8h;
typedef __attribute__((ext_vector_type(8)))  float    v8f;

// ---- WMMA helper: D = A(16x32 f16) * B(32x16 f16) + C(16x16 f32) ----
__device__ __forceinline__ v8f wmma16(v16h a, v16h b, v8f c) {
  return __builtin_amdgcn_wmma_f32_16x16x32_f16(false, a, false, b, (short)0, c,
                                                false, false);
}

// Per-lane operand gather from a row-major f16 matrix.
// Caller passes p = base + (row0 + (lane&15))*ld + k0 + (lane>=16 ? 8 : 0).
// v16h elems 0..7 = p[0..7], elems 8..15 = p[16..23]  (matches ISA A/B layout).
__device__ __forceinline__ v16h load_tile_h(const _Float16* __restrict__ p) {
  v8h lo = *(const v8h*)p;
  v8h hi = *(const v8h*)(p + 16);
  v16h r;
#pragma unroll
  for (int i = 0; i < 8; i++) { r[i] = lo[i]; r[i + 8] = hi[i]; }
  return r;
}

// Same gather from an fp32 matrix with on-the-fly convert to f16.
__device__ __forceinline__ v16h load_tile_f(const float* __restrict__ p) {
  v16h r;
#pragma unroll
  for (int i = 0; i < 8; i++) {
    r[i]     = (_Float16)p[i];
    r[i + 8] = (_Float16)p[i + 16];
  }
  return r;
}

// =====================================================================
// Kernel 1: fused Q/K/V projections.  y = X @ W^T + b, reshaped to heads.
// One wave computes a 16(M) x 64(N) block.  q,k -> [b,h,s,d] f16;
// v -> transposed [b,h,d,s] f16 so the AV GEMM's B-operand is contiguous.
// waves = 3 proj * 512 Mtiles * 8 Ntiles = 12288 -> 3072 blocks * 4 waves.
// =====================================================================
__global__ __launch_bounds__(128) void qkv_proj_kernel(
    const float* __restrict__ q_in, const float* __restrict__ k_in,
    const float* __restrict__ v_in,
    const float* __restrict__ Wq, const float* __restrict__ bq,
    const float* __restrict__ Wk, const float* __restrict__ bk,
    const float* __restrict__ Wv, const float* __restrict__ bv,
    _Float16* __restrict__ qh, _Float16* __restrict__ kh,
    _Float16* __restrict__ vt) {
  const int wave = (blockIdx.x << 2) + (threadIdx.x >> 5);
  const int lane = threadIdx.x & 31;
  const int proj = wave >> 12;          // 0=q 1=k 2=v
  const int rem  = wave & 4095;
  const int mt = rem >> 3, nt = rem & 7;

  const float* X    = (proj == 0) ? q_in : (proj == 1) ? k_in : v_in;
  const float* W    = (proj == 0) ? Wq   : (proj == 1) ? Wk   : Wv;
  const float* bias = (proj == 0) ? bq   : (proj == 1) ? bk   : bv;

  const int r   = lane & 15;
  const int kh8 = (lane >> 4) << 3;
  const int m0 = mt * 16, n0 = nt * 64;

  v8f acc[4] = {};
  for (int k0 = 0; k0 < DM; k0 += 32) {
    v16h a = load_tile_f(X + (size_t)(m0 + r) * DM + k0 + kh8);
#pragma unroll
    for (int j = 0; j < 4; j++) {
      v16h b = load_tile_f(W + (size_t)(n0 + 16 * j + r) * DM + k0 + kh8);
      acc[j] = wmma16(a, b, acc[j]);
    }
  }

  const int mbase = m0 + ((lane >> 4) << 3);
  const int ncol  = lane & 15;
#pragma unroll
  for (int j = 0; j < 4; j++) {
    const int n = n0 + 16 * j + ncol;
    const int h = n >> 6, d = n & 63;
    const float bv_ = bias[n];
#pragma unroll
    for (int i = 0; i < 8; i++) {
      const int m = mbase + i;
      const int bb = m >> 11, s = m & (SEQ - 1);
      const float val = acc[j][i] + bv_;
      if (proj == 2) {
        vt[((size_t)(bb * HEADS + h) * DHEAD + d) * SEQ + s] = (_Float16)val;
      } else {
        _Float16* dst = (proj == 0) ? qh : kh;
        dst[((size_t)(bb * HEADS + h) * SEQ + s) * DHEAD + d] = (_Float16)val;
      }
    }
  }
}

// =====================================================================
// Kernel 2: scores = (Q K^T)/8 + mask, row softmax, write att (fp32).
// Block = 256 thr (8 waves) owns a 16-row stripe of one (b,h):
//   LDS holds 16 x 2048 fp32 scores (128 KB, fits 320 KB WGP LDS).
// Each wave computes 16 of the 128 column tiles with 2 WMMAs each.
// =====================================================================
__global__ __launch_bounds__(256) void attn_softmax_kernel(
    const _Float16* __restrict__ qh, const _Float16* __restrict__ kh,
    const float* __restrict__ mask, float* __restrict__ att) {
  extern __shared__ char smem_raw[];
  float* sc     = (float*)smem_raw;            // [16][SEQ]
  float* redmax = sc + 16 * SEQ;               // [256]
  float* redsum = redmax + 256;                // [256]

  const int bh = blockIdx.y;                   // 0..31  (b*H + h)
  const int m0 = blockIdx.x * 16;
  const int wv = threadIdx.x >> 5;
  const int lane = threadIdx.x & 31;
  const int r = lane & 15;
  const int kh8 = (lane >> 4) << 3;

  const _Float16* qb = qh + (size_t)bh * SEQ * DHEAD;
  const _Float16* kb = kh + (size_t)bh * SEQ * DHEAD;

  // Q stripe operands are reused across all 16 column tiles of this wave.
  const v16h a0 = load_tile_h(qb + (size_t)(m0 + r) * DHEAD + 0  + kh8);
  const v16h a1 = load_tile_h(qb + (size_t)(m0 + r) * DHEAD + 32 + kh8);

  for (int t = wv; t < SEQ / 16; t += 8) {
    const int n0 = t * 16;
    v16h b0 = load_tile_h(kb + (size_t)(n0 + r) * DHEAD + 0  + kh8);
    v16h b1 = load_tile_h(kb + (size_t)(n0 + r) * DHEAD + 32 + kh8);
    v8f c = {};
    c = wmma16(a0, b0, c);
    c = wmma16(a1, b1, c);
    const int row = (lane >> 4) << 3;
    const int col = n0 + (lane & 15);
#pragma unroll
    for (int i = 0; i < 8; i++) sc[(row + i) * SEQ + col] = c[i] * 0.125f;
  }
  __syncthreads();

  // Row softmax: 16 threads cooperate per row.
  const int rrow = threadIdx.x >> 4;           // 0..15
  const int part = threadIdx.x & 15;           // 0..15
  const float* mrow = mask + (size_t)(m0 + rrow) * SEQ;

  float mx = -3.0e38f;
  for (int i = part; i < SEQ; i += 16) {
    float v = sc[rrow * SEQ + i] + mrow[i];
    sc[rrow * SEQ + i] = v;
    mx = fmaxf(mx, v);
  }
  redmax[threadIdx.x] = mx;
  __syncthreads();
  float m_all = -3.0e38f;
#pragma unroll
  for (int i = 0; i < 16; i++) m_all = fmaxf(m_all, redmax[rrow * 16 + i]);

  float sum = 0.0f;
  for (int i = part; i < SEQ; i += 16) {
    float e = __expf(sc[rrow * SEQ + i] - m_all);
    sc[rrow * SEQ + i] = e;
    sum += e;
  }
  redsum[threadIdx.x] = sum;
  __syncthreads();
  float s_all = 0.0f;
#pragma unroll
  for (int i = 0; i < 16; i++) s_all += redsum[rrow * 16 + i];
  const float inv = 1.0f / s_all;

  float* arow = att + ((size_t)bh * SEQ + m0 + rrow) * SEQ;
  for (int i = part; i < SEQ; i += 16) arow[i] = sc[rrow * SEQ + i] * inv;
}

// =====================================================================
// Kernel 3: O_heads = att @ V.  One wave: 16(M query rows) x 64(N=d).
// A from fp32 att (convert to f16 in regs), B from pre-transposed vt.
// waves = 32 (b,h) * 128 Mtiles = 4096 -> 1024 blocks * 4 waves.
// =====================================================================
__global__ __launch_bounds__(128) void attn_av_kernel(
    const float* __restrict__ att, const _Float16* __restrict__ vt,
    _Float16* __restrict__ oh) {
  const int wave = (blockIdx.x << 2) + (threadIdx.x >> 5);
  const int lane = threadIdx.x & 31;
  const int bh = wave >> 7;
  const int mt = wave & 127;
  const int m0 = mt * 16;
  const int r = lane & 15;
  const int kh8 = (lane >> 4) << 3;

  const float*    ab = att + (size_t)bh * SEQ * SEQ;
  const _Float16* vb = vt  + (size_t)bh * DHEAD * SEQ;

  v8f acc[4] = {};
  for (int k0 = 0; k0 < SEQ; k0 += 32) {
    const float* ap = ab + (size_t)(m0 + r) * SEQ + k0 + kh8;
    __builtin_prefetch(ap + 64, 0, 1);            // next att chunk (dominant stream)
    v16h a = load_tile_f(ap);
#pragma unroll
    for (int j = 0; j < 4; j++) {
      v16h b = load_tile_h(vb + (size_t)(16 * j + r) * SEQ + k0 + kh8);
      acc[j] = wmma16(a, b, acc[j]);
    }
  }

  const int b_ = bh >> 3, h = bh & 7;
  const int mbase = m0 + ((lane >> 4) << 3);
#pragma unroll
  for (int j = 0; j < 4; j++) {
    const int d = 16 * j + (lane & 15);
#pragma unroll
    for (int i = 0; i < 8; i++) {
      const int s = mbase + i;
      oh[(size_t)(b_ * SEQ + s) * DM + h * DHEAD + d] = (_Float16)acc[j][i];
    }
  }
}

// =====================================================================
// Kernel 4: out = O_heads @ Wo^T + bo (fp32 result).
// waves = 512 Mtiles * 8 Ntiles = 4096 -> 1024 blocks * 4 waves.
// =====================================================================
__global__ __launch_bounds__(128) void out_proj_kernel(
    const _Float16* __restrict__ oh, const float* __restrict__ Wo,
    const float* __restrict__ bo, float* __restrict__ out) {
  const int wave = (blockIdx.x << 2) + (threadIdx.x >> 5);
  const int lane = threadIdx.x & 31;
  const int mt = wave >> 3, nt = wave & 7;
  const int m0 = mt * 16, n0 = nt * 64;
  const int r = lane & 15;
  const int kh8 = (lane >> 4) << 3;

  v8f acc[4] = {};
  for (int k0 = 0; k0 < DM; k0 += 32) {
    v16h a = load_tile_h(oh + (size_t)(m0 + r) * DM + k0 + kh8);
#pragma unroll
    for (int j = 0; j < 4; j++) {
      v16h b = load_tile_f(Wo + (size_t)(n0 + 16 * j + r) * DM + k0 + kh8);
      acc[j] = wmma16(a, b, acc[j]);
    }
  }

  const int mbase = m0 + ((lane >> 4) << 3);
#pragma unroll
  for (int j = 0; j < 4; j++) {
    const int n = n0 + 16 * j + (lane & 15);
    const float bv_ = bo[n];
#pragma unroll
    for (int i = 0; i < 8; i++) {
      out[(size_t)(mbase + i) * DM + n] = acc[j][i] + bv_;
    }
  }
}

extern "C" void kernel_launch(void* const* d_in, const int* in_sizes, int n_in,
                              void* d_out, int out_size, void* d_ws,
                              size_t ws_size, hipStream_t stream) {
  (void)in_sizes; (void)n_in; (void)out_size; (void)ws_size;
  const float* q_in = (const float*)d_in[0];
  const float* k_in = (const float*)d_in[1];
  const float* v_in = (const float*)d_in[2];
  const float* mask = (const float*)d_in[3];
  const float* Wq = (const float*)d_in[4];  const float* bq = (const float*)d_in[5];
  const float* Wk = (const float*)d_in[6];  const float* bk = (const float*)d_in[7];
  const float* Wv = (const float*)d_in[8];  const float* bv = (const float*)d_in[9];
  const float* Wo = (const float*)d_in[10]; const float* bo = (const float*)d_in[11];

  float* out = (float*)d_out;              // [B,S,DM] fp32
  float* att = out + ATT_OFF;              // [B,H,S,S] fp32

  _Float16* qh = (_Float16*)d_ws;          // [B,H,S,64]  f16  (8 MB)
  _Float16* kh = qh + HSD;                 // [B,H,S,64]  f16  (8 MB)
  _Float16* vt = kh + HSD;                 // [B,H,64,S]  f16  (8 MB)
  _Float16* oh = vt + HSD;                 // [B,S,512]   f16  (8 MB)

  qkv_proj_kernel<<<3072, 128, 0, stream>>>(q_in, k_in, v_in, Wq, bq, Wk, bk,
                                            Wv, bv, qh, kh, vt);

  const size_t smem = (size_t)16 * SEQ * sizeof(float) + 2 * 256 * sizeof(float);
  attn_softmax_kernel<<<dim3(SEQ / 16, BATCH * HEADS), 256, smem, stream>>>(
      qh, kh, mask, att);

  attn_av_kernel<<<1024, 128, 0, stream>>>(att, vt, oh);

  out_proj_kernel<<<1024, 128, 0, stream>>>(oh, Wo, bo, out);
}
